// RoPETransformer_13151189860449
// MI455X (gfx1250) — matrix operand order repaired and checked by reading, not compiled, not run on hardware
//
#include <hip/hip_runtime.h>
#include <hip/hip_bf16.h>

// ---------------- model constants (match reference) ----------------
#define D_   768
#define H_   12
#define DK_  64
#define L_   512
#define B_   4
#define F_   3072
#define V_   32000
#define NL_  6
#define MT_  (B_*L_)   // 2048 tokens per side
#define BH_  (B_*H_)   // 48 attention batches

typedef __attribute__((ext_vector_type(16))) __bf16 v16bf;
typedef __attribute__((ext_vector_type(8)))  __bf16 v8bf;
typedef __attribute__((ext_vector_type(8)))  float  v8f;

__device__ __forceinline__ v16bf bf_combine(v8bf lo, v8bf hi) {
  v16bf o;
  #pragma unroll
  for (int i = 0; i < 8; ++i) { o[i] = lo[i]; o[i+8] = hi[i]; }
  return o;
}

// A 16x32 fragment (CDNA5 ISA 7.12.2): lane r,g holds row m=r,
// K = g*8+i (i<8) and K = 16+g*8+(i-8)
__device__ __forceinline__ v16bf ld_afrag(const __bf16* arow, int k, int g) {
  v8bf lo = *(const v8bf*)(arow + k + g*8);
  v8bf hi = *(const v8bf*)(arow + k + 16 + g*8);
  return bf_combine(lo, hi);
}
// B 32x16 fragment fed from Bt rows: lane r holds col n=r, K = g*8+i (i<8) and K = 16+g*8+(i-8)
__device__ __forceinline__ v16bf ld_bfrag(const __bf16* brow, int k, int g) {
  v8bf lo = *(const v8bf*)(brow + k + g*8);
  v8bf hi = *(const v8bf*)(brow + k + 16 + g*8);
  return bf_combine(lo, hi);
}
__device__ __forceinline__ v8f wmma_bf16(v16bf a, v16bf b, v8f c) {
  return __builtin_amdgcn_wmma_f32_16x16x32_bf16(
      false, a, false, b, (short)0, c, false, false);
}

// =====================================================================
// Batched WMMA GEMM:  C[z][M,N] = A[z][M,K] * Bt[z][N,K]^T (+bias)(+relu)
// A, Bt bf16 row-major; C fp32.  Register-blocked: each wave owns a
// 16(M) x 64(N) strip = 4 accumulators, so the A fragment is reused 4x
// per K-step (4 WMMAs / 160 B per lane ~ 13 flop/B from L2).
// 8 waves per 256-thread block cover 512 consecutive N.
// C batch base = C + (z/hDiv)*sCb + (z%hDiv)*sCh  (lets per-head output
// scatter straight into the merged [B*L, D] activation).
// Requires: M % 16 == 0, N % 64 == 0, K % 32 == 0 (true for all uses).
// =====================================================================
__global__ void __launch_bounds__(256) wmma_gemm_bf16(
    const __bf16* __restrict__ A,  long long sA, int lda,
    const __bf16* __restrict__ Bt, long long sB, int ldb,
    float* __restrict__ C, long long sCb, long long sCh, int hDiv, int ldc,
    const float* __restrict__ bias, int M, int N, int K, int relu)
{
  int z = blockIdx.z;
  const __bf16* Az = A  + (long long)z * sA;
  const __bf16* Bz = Bt + (long long)z * sB;
  float* Cz = C + (long long)(z / hDiv) * sCb + (long long)(z % hDiv) * sCh;

  int wave = threadIdx.x >> 5;
  int lane = threadIdx.x & 31;
  int tm = blockIdx.y << 4;
  int tn = ((blockIdx.x << 3) + wave) << 6;    // 64-wide N strip per wave
  if (tn >= N) return;                         // wave-uniform -> EXEC all-1s inside

  int r = lane & 15, g = lane >> 4;
  const __bf16* arow  = Az + (long long)(tm + r) * lda;
  const __bf16* brow0 = Bz + (long long)(tn      + r) * ldb;
  const __bf16* brow1 = brow0 + (long long)16 * ldb;
  const __bf16* brow2 = brow0 + (long long)32 * ldb;
  const __bf16* brow3 = brow0 + (long long)48 * ldb;

  v8f acc0 = {}, acc1 = {}, acc2 = {}, acc3 = {};
  #pragma unroll 2
  for (int k = 0; k < K; k += 32) {
    if (k + 128 < K) {                 // stream-ahead prefetch (global_prefetch_b8)
      __builtin_prefetch(arow  + k + 128, 0, 1);
      __builtin_prefetch(brow0 + k + 128, 0, 1);
      __builtin_prefetch(brow1 + k + 128, 0, 1);
      __builtin_prefetch(brow2 + k + 128, 0, 1);
      __builtin_prefetch(brow3 + k + 128, 0, 1);
    }
    v16bf av  = ld_afrag(arow,  k, g);
    v16bf bv0 = ld_bfrag(brow0, k, g);
    v16bf bv1 = ld_bfrag(brow1, k, g);
    v16bf bv2 = ld_bfrag(brow2, k, g);
    v16bf bv3 = ld_bfrag(brow3, k, g);
    acc0 = wmma_bf16(av, bv0, acc0);
    acc1 = wmma_bf16(av, bv1, acc1);
    acc2 = wmma_bf16(av, bv2, acc2);
    acc3 = wmma_bf16(av, bv3, acc3);
  }

  v8f accs[4] = {acc0, acc1, acc2, acc3};
  #pragma unroll
  for (int t = 0; t < 4; ++t) {
    int col = tn + t*16 + r;
    float bb = bias ? bias[col] : 0.0f;
    #pragma unroll
    for (int i = 0; i < 8; ++i) {
      float v = accs[t][i] + bb;
      if (relu) v = fmaxf(v, 0.0f);
      Cz[(long long)(tm + i + 8*g) * ldc + col] = v;
    }
  }
}

// ---------------- embedding gather: out[t, :] = tab[tok[t], :] -------
__global__ void embed_kernel(const int* __restrict__ tok,
                             const float* __restrict__ tab,
                             float* __restrict__ out, long long total)
{
  long long i = (long long)blockIdx.x * blockDim.x + threadIdx.x;
  if (i >= total) return;
  long long t = i / D_, d = i % D_;
  out[i] = tab[(long long)tok[t] * D_ + d];
}

// ---------------- fp32 -> bf16 convert -------------------------------
__global__ void f32_to_bf16_kernel(const float* __restrict__ in,
                                   __bf16* __restrict__ out, long long n)
{
  long long i = (long long)blockIdx.x * blockDim.x + threadIdx.x;
  if (i < n) out[i] = (__bf16)in[i];
}

// ---------------- LDS-tiled transpose: f32[K,N] -> bf16[N,K] ---------
__global__ void __launch_bounds__(256) transpose_f32_bf16(
    const float* __restrict__ in, __bf16* __restrict__ out, int K, int N)
{
  __shared__ float tile[32][33];
  int n0 = blockIdx.x * 32, k0 = blockIdx.y * 32;
  #pragma unroll
  for (int j = 0; j < 4; ++j) {
    int row = k0 + threadIdx.y + j*8, col = n0 + threadIdx.x;
    if (row < K && col < N)
      tile[threadIdx.y + j*8][threadIdx.x] = in[(long long)row * N + col];
  }
  __syncthreads();
  #pragma unroll
  for (int j = 0; j < 4; ++j) {
    int row = n0 + threadIdx.y + j*8, col = k0 + threadIdx.x;
    if (row < N && col < K)
      out[(long long)row * K + col] = (__bf16)tile[threadIdx.x][threadIdx.y + j*8];
  }
}

// ---------------- RoPE + head split: f32[B*L, D] -> bf16[BH, L, DK] --
__global__ void rope_pack_kernel(const float* __restrict__ x,
                                 __bf16* __restrict__ out)
{
  long long total = (long long)BH_ * L_ * (DK_/2);
  long long i = (long long)blockIdx.x * blockDim.x + threadIdx.x;
  if (i >= total) return;
  int p = (int)(i % (DK_/2));
  long long rest = i / (DK_/2);
  int l  = (int)(rest % L_);
  int bh = (int)(rest / L_);
  int h = bh % H_, b = bh / H_;
  const float* s = x + ((long long)(b*L_ + l) * D_ + h*DK_ + 2*p);
  float x0 = s[0], x1 = s[1];
  float freq = powf(10000.0f, -(2.0f * p) / (float)DK_);
  float ang = (float)l * freq;
  float sn = sinf(ang), cs = cosf(ang);
  __bf16* d = out + (((long long)bh * L_ + l) * DK_ + 2*p);
  d[0] = (__bf16)(x0 * cs - x1 * sn);
  d[1] = (__bf16)(x1 * cs + x0 * sn);
}

// ---------------- V pack (transposed): f32[B*L,D] -> bf16[BH, DK, L] -
__global__ void packv_kernel(const float* __restrict__ x,
                             __bf16* __restrict__ out)
{
  long long total = (long long)BH_ * DK_ * L_;
  long long i = (long long)blockIdx.x * blockDim.x + threadIdx.x;
  if (i >= total) return;
  int l = (int)(i % L_);
  long long rest = i / L_;
  int d  = (int)(rest % DK_);
  int bh = (int)(rest / DK_);
  int h = bh % H_, b = bh / H_;
  out[i] = (__bf16)x[(long long)(b*L_ + l) * D_ + h*DK_ + d];
}

// ---------------- scale + mask + softmax -> bf16 P -------------------
// grid = (L, BH), 256 threads, 2 elems/thread (Lk = 512)
__global__ void __launch_bounds__(256) softmax_kernel(
    const float* __restrict__ S, const int* __restrict__ mask,
    __bf16* __restrict__ P, int causal)
{
  __shared__ float red[256];
  int q = blockIdx.x, z = blockIdx.y, b = z / H_;
  const float* row = S + ((long long)z * L_ + q) * L_;
  __bf16* prow = P + ((long long)z * L_ + q) * L_;
  const int* mrow = causal ? (mask + ((long long)b * L_ + q) * L_)
                           : (mask + (long long)b * L_);
  float vals[2], mx = -3.4e38f;
  #pragma unroll
  for (int j = 0; j < 2; ++j) {
    int k = threadIdx.x + j*256;
    float v = row[k] * 0.125f;          // 1/sqrt(DK=64)
    if (mrow[k] == 0) v = -1.0e9f;
    vals[j] = v; mx = fmaxf(mx, v);
  }
  red[threadIdx.x] = mx; __syncthreads();
  for (int t = 128; t > 0; t >>= 1) {
    if (threadIdx.x < t) red[threadIdx.x] = fmaxf(red[threadIdx.x], red[threadIdx.x + t]);
    __syncthreads();
  }
  mx = red[0]; __syncthreads();
  float sum = 0.f;
  #pragma unroll
  for (int j = 0; j < 2; ++j) { vals[j] = __expf(vals[j] - mx); sum += vals[j]; }
  red[threadIdx.x] = sum; __syncthreads();
  for (int t = 128; t > 0; t >>= 1) {
    if (threadIdx.x < t) red[threadIdx.x] += red[threadIdx.x + t];
    __syncthreads();
  }
  float inv = 1.0f / red[0];
  #pragma unroll
  for (int j = 0; j < 2; ++j)
    prow[threadIdx.x + j*256] = (__bf16)(vals[j] * inv);
}

// ---------------- x = LayerNorm(x + a) * g + b ----------------------
// grid = MT rows, 256 threads, 3 elems/thread (D = 768)
__global__ void __launch_bounds__(256) ln_residual_kernel(
    float* __restrict__ x, const float* __restrict__ a,
    const float* __restrict__ g, const float* __restrict__ bt)
{
  __shared__ float red[256];
  long long r = blockIdx.x;
  float* xr = x + r * D_;
  const float* ar = a + r * D_;
  float v[3]; float s = 0.f;
  #pragma unroll
  for (int j = 0; j < 3; ++j) {
    int d = threadIdx.x + j*256;
    v[j] = xr[d] + ar[d]; s += v[j];
  }
  red[threadIdx.x] = s; __syncthreads();
  for (int t = 128; t > 0; t >>= 1) {
    if (threadIdx.x < t) red[threadIdx.x] += red[threadIdx.x + t];
    __syncthreads();
  }
  float mu = red[0] * (1.0f / D_); __syncthreads();
  float sq = 0.f;
  #pragma unroll
  for (int j = 0; j < 3; ++j) { float dl = v[j] - mu; sq += dl * dl; }
  red[threadIdx.x] = sq; __syncthreads();
  for (int t = 128; t > 0; t >>= 1) {
    if (threadIdx.x < t) red[threadIdx.x] += red[threadIdx.x + t];
    __syncthreads();
  }
  float inv = rsqrtf(red[0] * (1.0f / D_) + 1e-6f);
  #pragma unroll
  for (int j = 0; j < 3; ++j) {
    int d = threadIdx.x + j*256;
    xr[d] = (v[j] - mu) * inv * g[d] + bt[d];
  }
}

// =====================================================================
extern "C" void kernel_launch(void* const* d_in, const int* in_sizes, int n_in,
                              void* d_out, int out_size, void* d_ws, size_t ws_size,
                              hipStream_t stream)
{
  (void)in_sizes; (void)n_in; (void)out_size; (void)ws_size;

  // ---- inputs (JAX pytree flatten order: dicts by sorted key) ----
  const int*  src      = (const int*)d_in[0];
  const int*  trg      = (const int*)d_in[1];
  const int*  src_mask = (const int*)d_in[2];
  const int*  tgt_mask = (const int*)d_in[3];
  const float* d_b1  = (const float*)d_in[4];
  const float* d_b2  = (const float*)d_in[5];
  const float* d_b3  = (const float*)d_in[6];
  const float* d_ckb = (const float*)d_in[7];
  const float* d_ckw = (const float*)d_in[8];
  const float* d_cpb = (const float*)d_in[9];
  const float* d_cpw = (const float*)d_in[10];
  const float* d_cqb = (const float*)d_in[11];
  const float* d_cqw = (const float*)d_in[12];
  const float* d_cvb = (const float*)d_in[13];
  const float* d_cvw = (const float*)d_in[14];
  const float* d_f1b = (const float*)d_in[15];
  const float* d_f1w = (const float*)d_in[16];
  const float* d_f2b = (const float*)d_in[17];
  const float* d_f2w = (const float*)d_in[18];
  const float* d_g1  = (const float*)d_in[19];
  const float* d_g2  = (const float*)d_in[20];
  const float* d_g3  = (const float*)d_in[21];
  const float* d_skb = (const float*)d_in[22];
  const float* d_skw = (const float*)d_in[23];
  const float* d_spb = (const float*)d_in[24];
  const float* d_spw = (const float*)d_in[25];
  const float* d_sqb = (const float*)d_in[26];
  const float* d_sqw = (const float*)d_in[27];
  const float* d_svb = (const float*)d_in[28];
  const float* d_svw = (const float*)d_in[29];
  const float* dec_embed = (const float*)d_in[30];
  const float* e_b1  = (const float*)d_in[31];
  const float* e_b2  = (const float*)d_in[32];
  const float* e_f1b = (const float*)d_in[33];
  const float* e_f1w = (const float*)d_in[34];
  const float* e_f2b = (const float*)d_in[35];
  const float* e_f2w = (const float*)d_in[36];
  const float* e_g1  = (const float*)d_in[37];
  const float* e_g2  = (const float*)d_in[38];
  const float* e_kb  = (const float*)d_in[39];
  const float* e_kw  = (const float*)d_in[40];
  const float* e_pb  = (const float*)d_in[41];
  const float* e_pw  = (const float*)d_in[42];
  const float* e_qb  = (const float*)d_in[43];
  const float* e_qw  = (const float*)d_in[44];
  const float* e_vb  = (const float*)d_in[45];
  const float* e_vw  = (const float*)d_in[46];
  const float* enc_embed = (const float*)d_in[47];
  const float* out_b = (const float*)d_in[48];
  const float* out_w = (const float*)d_in[49];

  // ---- workspace carve (~216 MB) ----
  char* wp = (char*)d_ws;
  auto carve = [&](size_t bytes) -> void* {
    void* p = (void*)wp; wp += (bytes + 255) & ~(size_t)255; return p;
  };
  const size_t actF = (size_t)MT_ * D_ * sizeof(float);
  float*  xbuf   = (float*) carve(actF);                               // enc activ / eo
  float*  ybuf   = (float*) carve(actF);                               // dec activ
  float*  qf     = (float*) carve(actF);                               // q proj / attn out
  float*  kf     = (float*) carve(actF);
  float*  vf     = (float*) carve(actF);
  float*  aout   = (float*) carve(actF);                               // sublayer output
  float*  hidden = (float*) carve((size_t)MT_ * F_ * sizeof(float));   // ffn hidden
  float*  scores = (float*) carve((size_t)BH_ * L_ * L_ * sizeof(float));
  __bf16* Pmat   = (__bf16*)carve((size_t)BH_ * L_ * L_ * sizeof(__bf16));
  __bf16* Qr     = (__bf16*)carve((size_t)BH_ * L_ * DK_ * sizeof(__bf16));
  __bf16* Kr     = (__bf16*)carve((size_t)BH_ * L_ * DK_ * sizeof(__bf16));
  __bf16* Vt     = (__bf16*)carve((size_t)BH_ * DK_ * L_ * sizeof(__bf16));
  __bf16* actb   = (__bf16*)carve((size_t)MT_ * F_ * sizeof(__bf16));  // bf16 A scratch
  __bf16* wtb    = (__bf16*)carve((size_t)V_ * D_ * sizeof(__bf16));   // bf16 W^T scratch

  // ---- launch helpers ----
  auto conv = [&](const float* in, __bf16* out, long long n) {
    f32_to_bf16_kernel<<<(unsigned)((n + 255) / 256), 256, 0, stream>>>(in, out, n);
  };
  auto transW = [&](const float* in, int K, int N) {   // f32[K,N] -> wtb bf16[N,K]
    dim3 g((N + 31) / 32, (K + 31) / 32);
    transpose_f32_bf16<<<g, dim3(32, 8), 0, stream>>>(in, wtb, K, N);
  };
  auto gemm = [&](const __bf16* A, long long sA, int lda,
                  const __bf16* Bt, long long sB, int ldb,
                  float* C, long long sCb, long long sCh, int hDiv, int ldc,
                  const float* bias, int M, int N, int K, int relu, int batch) {
    dim3 g((N / 64 + 7) / 8, M / 16, batch);
    wmma_gemm_bf16<<<g, 256, 0, stream>>>(A, sA, lda, Bt, sB, ldb,
                                          C, sCb, sCh, hDiv, ldc, bias, M, N, K, relu);
  };
  auto lnres = [&](float* x, const float* a, const float* g, const float* b) {
    ln_residual_kernel<<<MT_, 256, 0, stream>>>(x, a, g, b);
  };

  // ---- full MHA sublayer: aout = proj(softmax(mask(QKᵀ/√d))·V) ----
  auto attention = [&](const float* xq, const float* xkv,
                       const float* qw, const float* qb,
                       const float* kw, const float* kb,
                       const float* vw, const float* vb,
                       const float* pw, const float* pb,
                       const int* mask, int causal) {
    // projections (WMMA)
    conv(xq, actb, (long long)MT_ * D_);
    transW(qw, D_, D_);
    gemm(actb, 0, D_, wtb, 0, D_, qf, 0, 0, 1, D_, qb, MT_, D_, D_, 0, 1);
    conv(xkv, actb, (long long)MT_ * D_);
    transW(kw, D_, D_);
    gemm(actb, 0, D_, wtb, 0, D_, kf, 0, 0, 1, D_, kb, MT_, D_, D_, 0, 1);
    transW(vw, D_, D_);
    gemm(actb, 0, D_, wtb, 0, D_, vf, 0, 0, 1, D_, vb, MT_, D_, D_, 0, 1);
    // RoPE + head packing
    long long nr = (long long)BH_ * L_ * (DK_ / 2);
    rope_pack_kernel<<<(unsigned)((nr + 255) / 256), 256, 0, stream>>>(qf, Qr);
    rope_pack_kernel<<<(unsigned)((nr + 255) / 256), 256, 0, stream>>>(kf, Kr);
    long long nv = (long long)BH_ * DK_ * L_;
    packv_kernel<<<(unsigned)((nv + 255) / 256), 256, 0, stream>>>(vf, Vt);
    // scores = Qr · Krᵀ  (batched WMMA, K=DK)
    gemm(Qr, (long long)L_ * DK_, DK_, Kr, (long long)L_ * DK_, DK_,
         scores, (long long)L_ * L_, 0, 1, L_, nullptr, L_, L_, DK_, 0, BH_);
    // mask + softmax -> bf16 P
    softmax_kernel<<<dim3(L_, BH_), 256, 0, stream>>>(scores, mask, Pmat, causal);
    // attn = P · V  (batched WMMA, K=L); scatter heads into qf[B*L, D]
    gemm(Pmat, (long long)L_ * L_, L_, Vt, (long long)DK_ * L_, L_,
         qf, (long long)L_ * D_, DK_, H_, D_, nullptr, L_, DK_, L_, 0, BH_);
    // output projection
    conv(qf, actb, (long long)MT_ * D_);
    transW(pw, D_, D_);
    gemm(actb, 0, D_, wtb, 0, D_, aout, 0, 0, 1, D_, pb, MT_, D_, D_, 0, 1);
  };

  auto ffn = [&](const float* x, const float* w1, const float* b1,
                 const float* w2, const float* b2) {
    conv(x, actb, (long long)MT_ * D_);
    transW(w1, D_, F_);
    gemm(actb, 0, D_, wtb, 0, D_, hidden, 0, 0, 1, F_, b1, MT_, F_, D_, 1, 1);
    conv(hidden, actb, (long long)MT_ * F_);
    transW(w2, F_, D_);
    gemm(actb, 0, F_, wtb, 0, F_, aout, 0, 0, 1, D_, b2, MT_, D_, F_, 0, 1);
  };

  const size_t DD = (size_t)D_ * D_;
  const size_t DF = (size_t)D_ * F_;

  // ================= encoder =================
  embed_kernel<<<(unsigned)(((long long)MT_ * D_ + 255) / 256), 256, 0, stream>>>(
      src, enc_embed, xbuf, (long long)MT_ * D_);
  for (int i = 0; i < NL_; ++i) {
    attention(xbuf, xbuf,
              e_qw + i*DD, e_qb + i*D_, e_kw + i*DD, e_kb + i*D_,
              e_vw + i*DD, e_vb + i*D_, e_pw + i*DD, e_pb + i*D_,
              src_mask, 0);
    lnres(xbuf, aout, e_g1 + i*D_, e_b1 + i*D_);
    ffn(xbuf, e_f1w + i*DF, e_f1b + i*F_, e_f2w + i*DF, e_f2b + i*D_);
    lnres(xbuf, aout, e_g2 + i*D_, e_b2 + i*D_);
  }
  // xbuf is now encoder output (eo)

  // ================= decoder =================
  embed_kernel<<<(unsigned)(((long long)MT_ * D_ + 255) / 256), 256, 0, stream>>>(
      trg, dec_embed, ybuf, (long long)MT_ * D_);
  for (int i = 0; i < NL_; ++i) {
    // masked self-attention
    attention(ybuf, ybuf,
              d_sqw + i*DD, d_sqb + i*D_, d_skw + i*DD, d_skb + i*D_,
              d_svw + i*DD, d_svb + i*D_, d_spw + i*DD, d_spb + i*D_,
              tgt_mask, 1);
    lnres(ybuf, aout, d_g1 + i*D_, d_b1 + i*D_);
    // cross-attention (kv from encoder output)
    attention(ybuf, xbuf,
              d_cqw + i*DD, d_cqb + i*D_, d_ckw + i*DD, d_ckb + i*D_,
              d_cvw + i*DD, d_cvb + i*D_, d_cpw + i*DD, d_cpb + i*D_,
              src_mask, 0);
    lnres(ybuf, aout, d_g2 + i*D_, d_b2 + i*D_);
    // feed-forward
    ffn(ybuf, d_f1w + i*DF, d_f1b + i*F_, d_f2w + i*DF, d_f2b + i*D_);
    lnres(ybuf, aout, d_g3 + i*D_, d_b3 + i*D_);
  }

  // ================= vocab projection (HBM-bound) =================
  conv(ybuf, actb, (long long)MT_ * D_);
  transW(out_w, D_, V_);
  gemm(actb, 0, D_, wtb, 0, D_, (float*)d_out, 0, 0, 1, V_,
       out_b, MT_, V_, D_, 0, 1);
}
